// NW_82729660055734
// MI455X (gfx1250) — matrix operand order repaired and checked by reading, not compiled, not run on hardware
//
#include <hip/hip_runtime.h>

// ---------------------------------------------------------------------------
// Nadaraya-Watson kernel regression, fused flash-style, bf16 WMMA (gfx1250).
//   out[m] = sum_n exp(-||x_m - t_n||^2/2) * y_n / sum_n exp(-||x_m - t_n||^2/2)
// GEMM term dominates (137 GFLOP); whole problem (~40MB) is L2-resident, so
// the kernel is compute-bound -> bf16 WMMA (16x16x32) with f32 accumulation.
// ---------------------------------------------------------------------------

typedef __attribute__((ext_vector_type(16))) __bf16 v16bf;
typedef __attribute__((ext_vector_type(8)))  __bf16 v8bf;
typedef __attribute__((ext_vector_type(8)))  float  v8f;

#define M_ROWS   8192
#define N_ROWS   32768
#define D_DIM    256
#define NCHUNKS  8
#define NCHUNK_COLS (N_ROWS / NCHUNKS)   // 4096 train columns per grid.y slice
#define BLOCK_M  128                     // 8 waves * 16 rows each

// -1 / (2 * ln 2): exp(-d/2) == exp2(d * C)
#define EXP2_SCALE (-0.72134752044448170f)

// ---------------------------------------------------------------------------
// Prep: convert f32 -> bf16 (row major) + row norms. Train rows also pack
// (norm, y) into a float2 so the main epilogue does one b64 load per tile.
// One wave32 per row; each lane handles 8 contiguous elements (2x float4).
// ---------------------------------------------------------------------------
__global__ __launch_bounds__(256)
void nw_prep(const float* __restrict__ Xtest,
             const float* __restrict__ Xtrain,
             const float* __restrict__ ytrain,
             __bf16* __restrict__ Xt, __bf16* __restrict__ Xn,
             float* __restrict__ sqt, float2* __restrict__ sny) {
  const int wave = (int)((blockIdx.x * blockDim.x + threadIdx.x) >> 5);
  const int lane = (int)(threadIdx.x & 31);

  const float* src;
  __bf16* dst;
  int train_row = -1;
  float* nrm = nullptr;
  if (wave < M_ROWS) {
    src = Xtest + (size_t)wave * D_DIM;
    dst = Xt + (size_t)wave * D_DIM;
    nrm = sqt + wave;
  } else {
    train_row = wave - M_ROWS;
    if (train_row >= N_ROWS) return;
    src = Xtrain + (size_t)train_row * D_DIM;
    dst = Xn + (size_t)train_row * D_DIM;
  }

  const int k = lane * 8;
  const float4 f0 = *(const float4*)(src + k);
  const float4 f1 = *(const float4*)(src + k + 4);

  v8bf h;
  h[0] = (__bf16)f0.x; h[1] = (__bf16)f0.y; h[2] = (__bf16)f0.z; h[3] = (__bf16)f0.w;
  h[4] = (__bf16)f1.x; h[5] = (__bf16)f1.y; h[6] = (__bf16)f1.z; h[7] = (__bf16)f1.w;
  *(v8bf*)(dst + k) = h;

  float s = f0.x * f0.x + f0.y * f0.y + f0.z * f0.z + f0.w * f0.w
          + f1.x * f1.x + f1.y * f1.y + f1.z * f1.z + f1.w * f1.w;
#pragma unroll
  for (int m = 16; m >= 1; m >>= 1) s += __shfl_xor(s, m, 32);
  if (lane == 0) {
    if (train_row < 0) *nrm = s;
    else               sny[train_row] = make_float2(s, ytrain[train_row]);
  }
}

// ---------------------------------------------------------------------------
// Main fused pass. Each wave: 16 output rows; all K=256 of A resident in
// 64 VGPRs. Per iteration: TWO independent 16-col N-tiles (two WMMA chains
// sharing A) with EXPLICIT double-buffering of B across k-steps, so every
// WMMA's B operands were loaded one full k-step (4x b128 + 2 WMMAs) earlier.
// ---------------------------------------------------------------------------
__global__ __launch_bounds__(256)
void nw_main(const __bf16* __restrict__ Xt, const __bf16* __restrict__ Xn,
             const float* __restrict__ sqt, const float2* __restrict__ sny,
             float2* __restrict__ part) {
  const int lane  = (int)(threadIdx.x & 31);
  const int wave  = (int)(threadIdx.x >> 5);
  const int m0    = (int)blockIdx.x * BLOCK_M + wave * 16;
  const int chunk = (int)blockIdx.y;
  const int n_begin = chunk * NCHUNK_COLS;

  union FragA { v16bf v; uint4 q[2]; };
  union FragB { v16bf v; uint4 q[2]; };

  // A fragments, contiguous K layout (matches B):
  //   lanes 0-15 : row = lane,    K in {0..15}
  //   lanes16-31 : row = lane-16, K in {16..31}
  FragA a[8];
  {
    const int arow  = m0 + (lane & 15);
    const int khalf = (lane >> 4) * 16;
    const __bf16* ap = Xt + (size_t)arow * D_DIM + khalf;
#pragma unroll
    for (int kb = 0; kb < 8; ++kb) {
      a[kb].q[0] = *(const uint4*)(ap + kb * 32);
      a[kb].q[1] = *(const uint4*)(ap + kb * 32 + 8);
    }
  }

  // C/D layout: VGPR r, lanes 0-15 -> row m0+r; lanes 16-31 -> row m0+8+r.
  const int rbase = m0 + (lane >> 4) * 8;
  float sqr[8];
  {
    const float4 s0 = *(const float4*)(sqt + rbase);
    const float4 s1 = *(const float4*)(sqt + rbase + 4);
    sqr[0] = s0.x; sqr[1] = s0.y; sqr[2] = s0.z; sqr[3] = s0.w;
    sqr[4] = s1.x; sqr[5] = s1.y; sqr[6] = s1.z; sqr[7] = s1.w;
  }

  // num/den accumulate this lane's columns from BOTH tiles; the final
  // cross-lane reduction sums over all columns of the chunk regardless.
  float num[8], den[8];
#pragma unroll
  for (int r = 0; r < 8; ++r) { num[r] = 0.0f; den[r] = 0.0f; }

  const int ncol = lane & 15;        // B/C column owned by this lane
  const int kh16 = (lane >> 4) * 16; // B layout: lane-half selects K 0-15 / 16-31

  for (int n0 = n_begin; n0 < n_begin + NCHUNK_COLS; n0 += 32) {
    const __bf16* bp0 = Xn + (size_t)(n0 + ncol) * D_DIM + kh16;
    const __bf16* bp1 = bp0 + (size_t)16 * D_DIM;

    // Warm WGP$ with next iteration's B rows. Locality 3 -> nearest scope:
    // WGP-scope prefetch pulls into all cache levels (data is L2-resident).
    __builtin_prefetch(bp0 + (size_t)32 * D_DIM, 0, 3);
    __builtin_prefetch(bp1 + (size_t)32 * D_DIM, 0, 3);

    v8f c0 = {};
    v8f c1 = {};

    // Explicit double-buffer of B fragments across k-steps.
    FragB b0[2], b1[2];
    b0[0].q[0] = *(const uint4*)(bp0);
    b0[0].q[1] = *(const uint4*)(bp0 + 8);
    b1[0].q[0] = *(const uint4*)(bp1);
    b1[0].q[1] = *(const uint4*)(bp1 + 8);
#pragma unroll
    for (int kb = 0; kb < 8; ++kb) {
      const int cur = kb & 1;
      const int nxt = cur ^ 1;
      if (kb < 7) {
        b0[nxt].q[0] = *(const uint4*)(bp0 + (kb + 1) * 32);
        b0[nxt].q[1] = *(const uint4*)(bp0 + (kb + 1) * 32 + 8);
        b1[nxt].q[0] = *(const uint4*)(bp1 + (kb + 1) * 32);
        b1[nxt].q[1] = *(const uint4*)(bp1 + (kb + 1) * 32 + 8);
      }
      c0 = __builtin_amdgcn_wmma_f32_16x16x32_bf16(
          false, a[kb].v, false, b0[cur].v, (short)0, c0, false, false);
      c1 = __builtin_amdgcn_wmma_f32_16x16x32_bf16(
          false, a[kb].v, false, b1[cur].v, (short)0, c1, false, false);
    }

    const float2 t0 = sny[n0 + ncol];        // (||t||^2, y) for tile 0 column
    const float2 t1 = sny[n0 + 16 + ncol];   // (||t||^2, y) for tile 1 column
#pragma unroll
    for (int r = 0; r < 8; ++r) {
      float d0 = fmaxf(sqr[r] + t0.x - 2.0f * c0[r], 0.0f);
      float d1 = fmaxf(sqr[r] + t1.x - 2.0f * c1[r], 0.0f);
      const float w0 = __builtin_amdgcn_exp2f(d0 * EXP2_SCALE);  // exp(-d/2)
      const float w1 = __builtin_amdgcn_exp2f(d1 * EXP2_SCALE);
      num[r] += w0 * t0.y + w1 * t1.y;
      den[r] += w0 + w1;
    }
  }

  // Reduce across the 16 columns held by the lanes of each half-wave.
#pragma unroll
  for (int r = 0; r < 8; ++r) {
    float nu = num[r], de = den[r];
#pragma unroll
    for (int m = 8; m >= 1; m >>= 1) {
      nu += __shfl_xor(nu, m, 32);
      de += __shfl_xor(de, m, 32);
    }
    if (ncol == 0)
      part[(size_t)(rbase + r) * NCHUNKS + chunk] = make_float2(nu, de);
  }
}

// ---------------------------------------------------------------------------
// Final: fold the NCHUNKS partials per row and divide.
// ---------------------------------------------------------------------------
__global__ __launch_bounds__(256)
void nw_final(const float2* __restrict__ part, float* __restrict__ out) {
  const int m = (int)(blockIdx.x * blockDim.x + threadIdx.x);
  if (m >= M_ROWS) return;
  float nu = 0.0f, de = 0.0f;
#pragma unroll
  for (int c = 0; c < NCHUNKS; ++c) {
    const float2 p = part[(size_t)m * NCHUNKS + c];
    nu += p.x;
    de += p.y;
  }
  out[m] = nu / de;
}

extern "C" void kernel_launch(void* const* d_in, const int* in_sizes, int n_in,
                              void* d_out, int out_size, void* d_ws, size_t ws_size,
                              hipStream_t stream) {
  (void)in_sizes; (void)n_in; (void)out_size; (void)ws_size;
  const float* Xtest  = (const float*)d_in[0];   // [8192, 256] f32
  const float* Xtrain = (const float*)d_in[1];   // [32768, 256] f32
  const float* ytrain = (const float*)d_in[2];   // [32768] f32

  char* ws = (char*)d_ws;
  size_t off = 0;
  __bf16* Xt = (__bf16*)(ws + off); off += (size_t)M_ROWS * D_DIM * sizeof(__bf16);
  __bf16* Xn = (__bf16*)(ws + off); off += (size_t)N_ROWS * D_DIM * sizeof(__bf16);
  float*  sqt = (float*)(ws + off); off += (size_t)M_ROWS * sizeof(float);
  float2* sny = (float2*)(ws + off); off += (size_t)N_ROWS * sizeof(float2);
  float2* part = (float2*)(ws + off); off += (size_t)M_ROWS * NCHUNKS * sizeof(float2);

  // 1) convert + row norms (+ pack (norm,y) for train rows)
  const int total_waves = M_ROWS + N_ROWS;
  nw_prep<<<dim3((total_waves * 32 + 255) / 256), dim3(256), 0, stream>>>(
      Xtest, Xtrain, ytrain, Xt, Xn, sqt, sny);

  // 2) fused distance/exp/weighted-sum pass
  nw_main<<<dim3(M_ROWS / BLOCK_M, NCHUNKS), dim3(256), 0, stream>>>(
      Xt, Xn, sqt, sny, part);

  // 3) fold partials, divide
  nw_final<<<dim3((M_ROWS + 255) / 256), dim3(256), 0, stream>>>(
      part, (float*)d_out);
}
